// HierarchicalTreeLSTMs_13718125543917
// MI455X (gfx1250) — matrix operand order repaired and checked by reading, not compiled, not run on hardware
//
#include <hip/hip_runtime.h>
#include <math.h>

// ---------------------------------------------------------------------------
// Problem constants (from reference): N=16384, L=8, D=256, H=256, R=64
// ---------------------------------------------------------------------------
#define NN   16384
#define LL   8
#define DD   256
#define HH   256
#define RR   64
#define TM   64          // nodes per workgroup tile

typedef __attribute__((ext_vector_type(16))) __bf16 v16bf;
typedef __attribute__((ext_vector_type(8)))  float  v8f;

union FragBF {
    v16bf v;
    uint4 q[2];
};

__device__ __forceinline__ float sigmoidf_(float x) {
    return 1.0f / (1.0f + __expf(-x));
}

// ---------------------------------------------------------------------------
// f32 -> bf16 conversion (weights)
// ---------------------------------------------------------------------------
__global__ void cvt_f32_to_bf16(const float* __restrict__ src,
                                __bf16* __restrict__ dst, int n) {
    for (int i = blockIdx.x * blockDim.x + threadIdx.x; i < n;
         i += gridDim.x * blockDim.x)
        dst[i] = (__bf16)src[i];
}

// bias fuse: out = a + b
__global__ void bias_sum(const float* __restrict__ a, const float* __restrict__ b,
                         float* __restrict__ out, int n) {
    for (int i = blockIdx.x * blockDim.x + threadIdx.x; i < n;
         i += gridDim.x * blockDim.x)
        out[i] = a[i] + b[i];
}

// ---------------------------------------------------------------------------
// Fused LSTM over the full chain. One workgroup = 64 nodes.
// LDS: x tile (bf16, 32KB) + h double buffer (bf16, 64KB) + c (f32, 64KB).
// 8 waves: wave w -> node rows [16*(w&3), +16), hidden half (w>>2).
// Gate matrix computed as 16x16 WMMA tiles; all 4 gates of a hidden block are
// accumulated together so c/h update happens entirely on-chip.
// ---------------------------------------------------------------------------
__global__ __launch_bounds__(256) void lstm_kernel(
    const float* __restrict__ x,        // [N, L, D] f32
    const __bf16* __restrict__ Wih,     // [4H, D] bf16
    const __bf16* __restrict__ Whh,     // [4H, H] bf16
    const float* __restrict__ bsum,     // [4H] f32 (bih + bhh)
    __bf16* __restrict__ hout)          // [N, H] bf16
{
    __shared__ __align__(16) __bf16 x_lds[TM * DD];
    __shared__ __align__(16) __bf16 h_lds[2][TM * HH];
    __shared__ __align__(16) float  c_lds[TM * HH];

    const int tid   = threadIdx.x;
    const int lane  = tid & 31;
    const int w     = tid >> 5;      // wave 0..7
    const int msub  = w & 3;         // node sub-tile (16 rows each)
    const int jhalf = w >> 2;        // hidden half (8 blocks of 16 each)
    const int mbase = msub * 16;
    const int n16   = lane & 15;
    const int g     = lane >> 4;
    const int node0 = blockIdx.x * TM;

    // zero init h(t=0) and c
    for (int i = tid; i < TM * HH; i += 256) {
        h_lds[0][i] = (__bf16)0.0f;
        c_lds[i]    = 0.0f;
    }
    int cur = 0;
    __syncthreads();

    for (int t = 0; t < LL; ++t) {
        // cooperative load of x_t tile, f32 -> bf16 (coalesced along k)
        for (int i = tid; i < TM * DD; i += 256) {
            const int m = i >> 8, k = i & 255;
            x_lds[i] = (__bf16)x[(size_t)(node0 + m) * (LL * DD) + (size_t)t * DD + k];
        }
        // prefetch next step's x tile rows into cache
        if (t + 1 < LL) {
            const int m = tid >> 2;  // 64 rows, 4 threads/row
            __builtin_prefetch(&x[(size_t)(node0 + m) * (LL * DD) +
                                  (size_t)(t + 1) * DD + (tid & 3) * 64], 0, 1);
        }
        __syncthreads();
        const int nxt = cur ^ 1;

        for (int jj = 0; jj < 8; ++jj) {
            const int jb  = jhalf * 8 + jj;
            const int col = jb * 16 + n16;   // hidden unit / C-matrix column

            v8f acc[4];
            for (int q = 0; q < 4; ++q) {
                const float bv = bsum[q * HH + col];
                acc[q] = (v8f){bv, bv, bv, bv, bv, bv, bv, bv};
            }

            // x_t @ Wih^T  (K = D = 256, 8 chunks of 32)
            for (int kc = 0; kc < 8; ++kc) {
                const int kb = kc * 32;
                FragBF a;
                const __bf16* ar = &x_lds[(mbase + n16) * DD + kb];
                a.q[0] = *(const uint4*)(ar + g * 8);        // K = kb+g*8 .. +7
                a.q[1] = *(const uint4*)(ar + 16 + g * 8);   // K = kb+16+g*8 ..
                for (int q = 0; q < 4; ++q) {
                    FragBF b;
                    const __bf16* br = Wih + (size_t)(q * HH + col) * DD + kb;
                    b.q[0] = *(const uint4*)(br + g * 8);        // K = kb+g*8 .. +7
                    b.q[1] = *(const uint4*)(br + 16 + g * 8);   // K = kb+16+g*8 ..
                    acc[q] = __builtin_amdgcn_wmma_f32_16x16x32_bf16(
                        false, a.v, false, b.v, (short)0, acc[q], false, false);
                }
            }
            // h_{t-1} @ Whh^T  (K = H = 256)
            for (int kc = 0; kc < 8; ++kc) {
                const int kb = kc * 32;
                FragBF a;
                const __bf16* ar = &h_lds[cur][(mbase + n16) * HH + kb];
                a.q[0] = *(const uint4*)(ar + g * 8);
                a.q[1] = *(const uint4*)(ar + 16 + g * 8);
                for (int q = 0; q < 4; ++q) {
                    FragBF b;
                    const __bf16* br = Whh + (size_t)(q * HH + col) * HH + kb;
                    b.q[0] = *(const uint4*)(br + g * 8);
                    b.q[1] = *(const uint4*)(br + 16 + g * 8);
                    acc[q] = __builtin_amdgcn_wmma_f32_16x16x32_bf16(
                        false, a.v, false, b.v, (short)0, acc[q], false, false);
                }
            }

            // activations + cell/hidden update (each element owned by one wave)
            for (int r = 0; r < 8; ++r) {
                const int m   = mbase + r + g * 8;   // C/D layout: VGPR r -> row
                const int idx = m * HH + col;
                const float iv = sigmoidf_(acc[0][r]);
                const float fv = sigmoidf_(acc[1][r]);
                const float gv = tanhf(acc[2][r]);
                const float ov = sigmoidf_(acc[3][r]);
                const float cn = fv * c_lds[idx] + iv * gv;
                c_lds[idx]      = cn;
                h_lds[nxt][idx] = (__bf16)(ov * tanhf(cn));
            }
        }
        __syncthreads();
        cur ^= 1;
    }

    // write final hidden state (bf16) to workspace
    for (int i = tid; i < TM * HH; i += 256) {
        const int m = i >> 8, k = i & 255;
        hout[(size_t)(node0 + m) * HH + k] = h_lds[cur][i];
    }
}

// ---------------------------------------------------------------------------
// Encoder: out = tanh([hl|hr|rel] @ Wenc^T + benc).  K = 2H+R = 576.
// ---------------------------------------------------------------------------
#define KC 576
__global__ __launch_bounds__(256) void enc_kernel(
    const __bf16* __restrict__ hl, const __bf16* __restrict__ hr,
    const float* __restrict__ rel,       // [N, R] f32
    const __bf16* __restrict__ Wenc,     // [H, 576] bf16
    const float* __restrict__ benc,      // [H] f32
    float* __restrict__ out)             // [N, H] f32
{
    __shared__ __align__(16) __bf16 con[TM * KC];

    const int tid   = threadIdx.x;
    const int lane  = tid & 31;
    const int w     = tid >> 5;
    const int msub  = w & 3;
    const int jhalf = w >> 2;
    const int mbase = msub * 16;
    const int n16   = lane & 15;
    const int g     = lane >> 4;
    const int node0 = blockIdx.x * TM;

    // stage concat tile in LDS
    for (int i = tid; i < TM * HH; i += 256) {
        const int m = i >> 8, k = i & 255;
        con[m * KC + k]       = hl[(size_t)(node0 + m) * HH + k];
        con[m * KC + HH + k]  = hr[(size_t)(node0 + m) * HH + k];
    }
    for (int i = tid; i < TM * RR; i += 256) {
        const int m = i >> 6, k = i & 63;
        con[m * KC + 2 * HH + k] = (__bf16)rel[(size_t)(node0 + m) * RR + k];
    }
    __syncthreads();

    for (int jj = 0; jj < 8; ++jj) {
        const int jb  = jhalf * 8 + jj;
        const int col = jb * 16 + n16;
        const float bv = benc[col];
        v8f acc = (v8f){bv, bv, bv, bv, bv, bv, bv, bv};

        for (int kc = 0; kc < KC / 32; ++kc) {
            const int kb = kc * 32;
            FragBF a;
            const __bf16* ar = &con[(mbase + n16) * KC + kb];
            a.q[0] = *(const uint4*)(ar + g * 8);
            a.q[1] = *(const uint4*)(ar + 16 + g * 8);
            FragBF b;
            const __bf16* br = Wenc + (size_t)col * KC + kb;
            b.q[0] = *(const uint4*)(br + g * 8);
            b.q[1] = *(const uint4*)(br + 16 + g * 8);
            acc = __builtin_amdgcn_wmma_f32_16x16x32_bf16(
                false, a.v, false, b.v, (short)0, acc, false, false);
        }
        for (int r = 0; r < 8; ++r) {
            const int m = mbase + r + g * 8;
            out[(size_t)(node0 + m) * HH + col] = tanhf(acc[r]);
        }
    }
}

// ---------------------------------------------------------------------------
// Host-side orchestration
// ---------------------------------------------------------------------------
extern "C" void kernel_launch(void* const* d_in, const int* in_sizes, int n_in,
                              void* d_out, int out_size, void* d_ws, size_t ws_size,
                              hipStream_t stream) {
    (void)in_sizes; (void)n_in; (void)out_size; (void)ws_size;

    const float* left  = (const float*)d_in[0];
    const float* right = (const float*)d_in[1];
    const float* rel   = (const float*)d_in[2];
    const float* Wih_l = (const float*)d_in[3];
    const float* Whh_l = (const float*)d_in[4];
    const float* bih_l = (const float*)d_in[5];
    const float* bhh_l = (const float*)d_in[6];
    const float* Wih_r = (const float*)d_in[7];
    const float* Whh_r = (const float*)d_in[8];
    const float* bih_r = (const float*)d_in[9];
    const float* bhh_r = (const float*)d_in[10];
    const float* Wenc  = (const float*)d_in[11];
    const float* benc  = (const float*)d_in[12];
    float* out = (float*)d_out;

    // workspace layout (bytes)
    const size_t W_ELEMS    = 4 * HH * DD;     // 262144
    const size_t WENC_ELEMS = HH * (2 * HH + RR);  // 147456
    char* ws = (char*)d_ws;
    __bf16* wih_l_bf = (__bf16*)(ws);
    __bf16* whh_l_bf = (__bf16*)(ws + 524288);
    __bf16* wih_r_bf = (__bf16*)(ws + 1048576);
    __bf16* whh_r_bf = (__bf16*)(ws + 1572864);
    __bf16* wenc_bf  = (__bf16*)(ws + 2097152);
    float*  bsum_l   = (float*) (ws + 2392064);
    float*  bsum_r   = (float*) (ws + 2396160);
    __bf16* hl_bf    = (__bf16*)(ws + 2400256);
    __bf16* hr_bf    = (__bf16*)(ws + 2400256 + (size_t)NN * HH * 2);

    // weight conversion + bias fusion
    cvt_f32_to_bf16<<<512, 256, 0, stream>>>(Wih_l, wih_l_bf, (int)W_ELEMS);
    cvt_f32_to_bf16<<<512, 256, 0, stream>>>(Whh_l, whh_l_bf, (int)W_ELEMS);
    cvt_f32_to_bf16<<<512, 256, 0, stream>>>(Wih_r, wih_r_bf, (int)W_ELEMS);
    cvt_f32_to_bf16<<<512, 256, 0, stream>>>(Whh_r, whh_r_bf, (int)W_ELEMS);
    cvt_f32_to_bf16<<<512, 256, 0, stream>>>(Wenc,  wenc_bf,  (int)WENC_ELEMS);
    bias_sum<<<4, 256, 0, stream>>>(bih_l, bhh_l, bsum_l, 4 * HH);
    bias_sum<<<4, 256, 0, stream>>>(bih_r, bhh_r, bsum_r, 4 * HH);

    // the two chain LSTMs (independent)
    lstm_kernel<<<NN / TM, 256, 0, stream>>>(left,  wih_l_bf, whh_l_bf, bsum_l, hl_bf);
    lstm_kernel<<<NN / TM, 256, 0, stream>>>(right, wih_r_bf, whh_r_bf, bsum_r, hr_bf);

    // encoder
    enc_kernel<<<NN / TM, 256, 0, stream>>>(hl_bf, hr_bf, rel, wenc_bf, benc, out);
}
